// LlamaAttention_78589311582275
// MI455X (gfx1250) — matrix operand order repaired and checked by reading, not compiled, not run on hardware
//
#include <hip/hip_runtime.h>
#include <math.h>

typedef __attribute__((ext_vector_type(16))) __bf16 v16bf;
typedef __attribute__((ext_vector_type(8)))  __bf16 v8bf;
typedef __attribute__((ext_vector_type(8)))  float  v8f;

#define B_   2
#define S_   2048
#define E_   2048
#define H_   16
#define D_   128
#define MTOT (B_ * S_)   // 4096

// ---------------- WMMA + fragment helpers ----------------

__device__ __forceinline__ v8f wmma_bf16(v16bf a, v16bf b, v8f c) {
  return __builtin_amdgcn_wmma_f32_16x16x32_bf16(
      /*neg_a=*/false, a, /*neg_b=*/false, b,
      /*c_mod=*/(short)0, c, /*reuse_a=*/false, /*reuse_b=*/false);
}

// A-matrix 16x32 bf16 fragment: per-lane row = lane%16, khalf = lane/16.
// elements 0..15 at K = 16*khalf + e (contiguous)
__device__ __forceinline__ v16bf ld_afrag(const __bf16* base) {
  v8bf lo = *(const v8bf*)(base);
  v8bf hi = *(const v8bf*)(base + 8);
  v16bf r;
#pragma unroll
  for (int i = 0; i < 8; ++i) { r[i] = lo[i]; r[i + 8] = hi[i]; }
  return r;
}

// B-matrix 32x16 bf16 fragment: per-lane col = lane%16, khalf = lane/16.
// elements 0..15 at K = 16*khalf + e (contiguous)
__device__ __forceinline__ v16bf ld_bfrag(const __bf16* base) {
  v8bf lo = *(const v8bf*)(base);
  v8bf hi = *(const v8bf*)(base + 8);
  v16bf r;
#pragma unroll
  for (int i = 0; i < 8; ++i) { r[i] = lo[i]; r[i + 8] = hi[i]; }
  return r;
}

// ---------------- async global -> LDS copy (CDNA5) ----------------
// 16 bytes per lane; tracked with ASYNCcnt. LDS address = low 32 bits of the
// generic pointer (ISA 10.2, LDS aperture).
__device__ __forceinline__ void async_cp16(void* lds, const void* gptr) {
  asm volatile("global_load_async_to_lds_b128 %0, %1, off"
               :: "v"((uint32_t)(uintptr_t)lds), "v"(gptr)
               : "memory");
}
__device__ __forceinline__ void wait_async0() {
  asm volatile("s_wait_asynccnt 0" ::: "memory");
}

// ---------------- conversion kernels ----------------

__global__ void cvt_x_kernel(const float* __restrict__ x, __bf16* __restrict__ xb) {
  size_t i = (size_t)blockIdx.x * blockDim.x + threadIdx.x;
  xb[i] = (__bf16)x[i];
}

// wt[n][k] = w[k][n]   (2048 x 2048)
__global__ void cvt_wt_kernel(const float* __restrict__ w, __bf16* __restrict__ wt) {
  size_t i = (size_t)blockIdx.x * blockDim.x + threadIdx.x;
  int k = (int)(i & (E_ - 1));
  int n = (int)(i >> 11);
  wt[(size_t)n * E_ + k] = (__bf16)w[(size_t)k * E_ + n];
}

// ---------------- QKV GEMM + RoPE (LDS-tiled, async double-buffered) ----------
// grid (MTOT/64, H, 3), block 256 (8 waves). Workgroup tile: 64 rows x 128 cols
// (= all columns of head h). Wave w: row-tile rt=w/2, column tiles
// {c0, c0+1, c0+4, c0+5} with c0=(w&1)*2, so RoPE pairs (d, d+64) stay in-wave.
__global__ void __launch_bounds__(256)
qkv_rope_kernel(const __bf16* __restrict__ Xb,
                const __bf16* __restrict__ WqT,
                const __bf16* __restrict__ WkT,
                const __bf16* __restrict__ WvT,
                const float*  __restrict__ cosp,
                const float*  __restrict__ sinp,
                __bf16* __restrict__ Qb,   // [B,H,S,D]
                __bf16* __restrict__ Kb,   // [B,H,S,D]
                __bf16* __restrict__ Vt) { // [B,H,D,S]
  __shared__ __bf16 Ab[2][64][32];    // 2 x 4 KB
  __shared__ __bf16 Bb[2][128][32];   // 2 x 8 KB

  int tid  = threadIdx.x;
  int lane = tid & 31, wave = tid >> 5;
  int col  = lane & 15, half = lane >> 4;

  int m0    = blockIdx.x * 64;
  int h     = blockIdx.y;
  int which = blockIdx.z;                      // 0=Q 1=K 2=V
  const __bf16* WT = (which == 0) ? WqT : (which == 1) ? WkT : WvT;
  const __bf16* Wh = WT + (size_t)h * D_ * E_; // rows h*128 .. h*128+127

  auto stage = [&](int buf, int kk) {
    {   // A tile: 64 rows x 32 k = 4 KB = 256 lanes x 16 B
      int row = tid >> 2, seg = tid & 3;
      async_cp16(&Ab[buf][row][seg * 8], Xb + (size_t)(m0 + row) * E_ + kk + seg * 8);
    }
#pragma unroll
    for (int j = 0; j < 2; ++j) {  // B tile: 128 rows x 32 k = 8 KB
      int idx = tid + j * 256;
      int row = idx >> 2, seg = idx & 3;
      async_cp16(&Bb[buf][row][seg * 8], Wh + (size_t)row * E_ + kk + seg * 8);
    }
  };

  int rt = wave >> 1;
  int c0 = (wave & 1) * 2;
  const int cts[4] = {c0, c0 + 1, c0 + 4, c0 + 5};

  v8f acc[4];
#pragma unroll
  for (int j = 0; j < 4; ++j) acc[j] = (v8f){};

  stage(0, 0);
  for (int i = 0; i < E_ / 32; ++i) {
    int cur = i & 1;
    wait_async0();
    __syncthreads();
    if (i + 1 < E_ / 32) stage(cur ^ 1, (i + 1) * 32);

    v16bf a = ld_afrag(&Ab[cur][rt * 16 + col][16 * half]);
#pragma unroll
    for (int j = 0; j < 4; ++j) {
      v16bf b = ld_bfrag(&Bb[cur][cts[j] * 16 + col][16 * half]);
      acc[j] = wmma_bf16(a, b, acc[j]);
    }
  }

  // epilogue: RoPE pairs (acc[p], acc[p+2]) -> columns (d0, d0+64)
#pragma unroll
  for (int p = 0; p < 2; ++p) {
    v8f a0 = acc[p], a1 = acc[p + 2];
    int d0 = (c0 + p) * 16 + col;   // in [0,64)
    int d1 = d0 + 64;
#pragma unroll
    for (int r = 0; r < 8; ++r) {
      int mg = m0 + rt * 16 + r + 8 * half;  // global row in [B*S]
      int bb = mg >> 11;
      int s  = mg & (S_ - 1);
      int bh = bb * H_ + h;
      if (which == 2) {
        Vt[((size_t)bh * D_ + d0) * S_ + s] = (__bf16)a0[r];
        Vt[((size_t)bh * D_ + d1) * S_ + s] = (__bf16)a1[r];
      } else {
        float c0v = cosp[(size_t)s * D_ + d0];
        float s0v = sinp[(size_t)s * D_ + d0];
        float c1v = cosp[(size_t)s * D_ + d1];
        float s1v = sinp[(size_t)s * D_ + d1];
        float q0 = a0[r] * c0v - a1[r] * s0v;  // rotate_half: first half -x2*sin
        float q1 = a1[r] * c1v + a0[r] * s1v;  // second half +x1*sin
        __bf16* dst = (which == 0) ? Qb : Kb;
        dst[((size_t)bh * S_ + s) * D_ + d0] = (__bf16)q0;
        dst[((size_t)bh * S_ + s) * D_ + d1] = (__bf16)q1;
      }
    }
  }
}

// ---------------- flash attention (shared K/V tiles in LDS) ----------------
// grid (B*H, 16), block 256 (8 waves). Each wave owns one 16-query tile; the
// block shares each 32-key K chunk (32x128) and V^T chunk (128x32) via LDS.
__global__ void __launch_bounds__(256)
attn_kernel(const __bf16* __restrict__ Qb,  // [B,H,S,D]
            const __bf16* __restrict__ Kb,  // [B,H,S,D]
            const __bf16* __restrict__ Vt,  // [B,H,D,S]
            __bf16* __restrict__ AO) {      // [B,S,H*D]
  __shared__ __bf16 Ks[2][32][128];   // 2 x 8 KB
  __shared__ __bf16 Vs[2][128][32];   // 2 x 8 KB
  __shared__ __bf16 plds[8][16][32];  // per-wave P tile, 8 KB

  int tid  = threadIdx.x;
  int lane = tid & 31, wave = tid >> 5;
  int col  = lane & 15, half = lane >> 4;

  int bh = blockIdx.x;
  int b  = bh >> 4;
  int h  = bh & (H_ - 1);
  int qtile = blockIdx.y * 8 + wave;        // 0..127

  const __bf16* Qhead = Qb + (size_t)bh * S_ * D_;
  const __bf16* Khead = Kb + (size_t)bh * S_ * D_;
  const __bf16* Vhead = Vt + (size_t)bh * D_ * S_;

  auto stage = [&](int buf, int n0) {
#pragma unroll
    for (int j = 0; j < 2; ++j) {  // K chunk: 32 keys x 128 d = 8 KB
      int idx = tid + j * 256;
      int row = idx >> 4, seg = idx & 15;
      async_cp16(&Ks[buf][row][seg * 8], Khead + (size_t)(n0 + row) * D_ + seg * 8);
    }
#pragma unroll
    for (int j = 0; j < 2; ++j) {  // V^T chunk: 128 d x 32 keys = 8 KB
      int idx = tid + j * 256;
      int row = idx >> 2, seg = idx & 3;
      async_cp16(&Vs[buf][row][seg * 8], Vhead + (size_t)row * S_ + n0 + seg * 8);
    }
  };

  // Preload Q A-fragments (4 K-chunks of D=128)
  int qrow = qtile * 16 + col;
  v16bf aq[4];
#pragma unroll
  for (int c = 0; c < 4; ++c)
    aq[c] = ld_afrag(Qhead + (size_t)qrow * D_ + c * 32 + 16 * half);

  float m[8], l[8];
  v8f oacc[8];
#pragma unroll
  for (int r = 0; r < 8; ++r) { m[r] = -3.0e38f; l[r] = 0.0f; }
#pragma unroll
  for (int dt = 0; dt < 8; ++dt) oacc[dt] = (v8f){};

  stage(0, 0);
  for (int i = 0; i < S_ / 32; ++i) {
    int cur = i & 1;
    wait_async0();
    __syncthreads();
    if (i + 1 < S_ / 32) stage(cur ^ 1, (i + 1) * 32);

    // scores: two 16x16 tiles (keys 0..15 and 16..31 of this chunk)
    v8f sc0 = {}, sc1 = {};
#pragma unroll
    for (int c = 0; c < 4; ++c) {
      v16bf bk0 = ld_bfrag(&Ks[cur][col][c * 32 + 16 * half]);
      v16bf bk1 = ld_bfrag(&Ks[cur][16 + col][c * 32 + 16 * half]);
      sc0 = wmma_bf16(aq[c], bk0, sc0);
      sc1 = wmma_bf16(aq[c], bk1, sc1);
    }

    // online softmax per row (row M = r + 8*half; 16 lanes share a row)
    float scale[8];
#pragma unroll
    for (int r = 0; r < 8; ++r) {
      float mx = fmaxf(sc0[r], sc1[r]);
#pragma unroll
      for (int off = 8; off >= 1; off >>= 1)
        mx = fmaxf(mx, __shfl_xor(mx, off, 32));
      float mn = fmaxf(m[r], mx);
      float sc = __expf(m[r] - mn);
      float p0 = __expf(sc0[r] - mn);
      float p1 = __expf(sc1[r] - mn);
      float rs = p0 + p1;
#pragma unroll
      for (int off = 8; off >= 1; off >>= 1)
        rs += __shfl_xor(rs, off, 32);
      l[r] = l[r] * sc + rs;
      m[r] = mn;
      scale[r] = sc;
      plds[wave][r + 8 * half][col]      = (__bf16)p0;
      plds[wave][r + 8 * half][16 + col] = (__bf16)p1;
    }
#pragma unroll
    for (int dt = 0; dt < 8; ++dt)
#pragma unroll
      for (int r = 0; r < 8; ++r) oacc[dt][r] *= scale[r];

    asm volatile("s_wait_dscnt 0" ::: "memory");

    // O += P * V from LDS
    v16bf pa = ld_afrag(&plds[wave][col][16 * half]);
#pragma unroll
    for (int dt = 0; dt < 8; ++dt) {
      v16bf bv = ld_bfrag(&Vs[cur][dt * 16 + col][16 * half]);
      oacc[dt] = wmma_bf16(pa, bv, oacc[dt]);
    }
  }

  // finalize: divide by l, write [B,S,H*D] bf16
#pragma unroll
  for (int r = 0; r < 8; ++r) {
    float inv = 1.0f / l[r];
    int s = qtile * 16 + r + 8 * half;
    size_t rowbase = ((size_t)(b * S_ + s)) * (H_ * D_) + h * D_;
#pragma unroll
    for (int dt = 0; dt < 8; ++dt)
      AO[rowbase + dt * 16 + col] = (__bf16)(oacc[dt][r] * inv);
  }
}

// ---------------- output projection (LDS-tiled, async double-buffered) ------
// grid (MTOT/64, E/128), block 256. Workgroup tile: 64 rows x 128 cols.
__global__ void __launch_bounds__(256)
oproj_kernel(const __bf16* __restrict__ AO,   // [B*S, 2048] bf16
             const __bf16* __restrict__ WoT,  // [2048, 2048] transposed bf16
             float* __restrict__ out) {       // [B*S, 2048] f32
  __shared__ __bf16 Ab[2][64][32];
  __shared__ __bf16 Bb[2][128][32];

  int tid  = threadIdx.x;
  int lane = tid & 31, wave = tid >> 5;
  int col  = lane & 15, half = lane >> 4;

  int m0 = blockIdx.x * 64;
  int nb = blockIdx.y * 128;
  const __bf16* Wh = WoT + (size_t)nb * E_;

  auto stage = [&](int buf, int kk) {
    { int row = tid >> 2, seg = tid & 3;
      async_cp16(&Ab[buf][row][seg * 8], AO + (size_t)(m0 + row) * E_ + kk + seg * 8); }
#pragma unroll
    for (int j = 0; j < 2; ++j) {
      int idx = tid + j * 256;
      int row = idx >> 2, seg = idx & 3;
      async_cp16(&Bb[buf][row][seg * 8], Wh + (size_t)row * E_ + kk + seg * 8);
    }
  };

  int rt = wave >> 1;
  int c0 = (wave & 1) * 2;
  const int cts[4] = {c0, c0 + 1, c0 + 4, c0 + 5};

  v8f acc[4];
#pragma unroll
  for (int j = 0; j < 4; ++j) acc[j] = (v8f){};

  stage(0, 0);
  for (int i = 0; i < E_ / 32; ++i) {
    int cur = i & 1;
    wait_async0();
    __syncthreads();
    if (i + 1 < E_ / 32) stage(cur ^ 1, (i + 1) * 32);

    v16bf a = ld_afrag(&Ab[cur][rt * 16 + col][16 * half]);
#pragma unroll
    for (int j = 0; j < 4; ++j) {
      v16bf b = ld_bfrag(&Bb[cur][cts[j] * 16 + col][16 * half]);
      acc[j] = wmma_bf16(a, b, acc[j]);
    }
  }

#pragma unroll
  for (int j = 0; j < 4; ++j)
#pragma unroll
    for (int r = 0; r < 8; ++r)
      out[(size_t)(m0 + rt * 16 + r + 8 * half) * E_ + nb + cts[j] * 16 + col] = acc[j][r];
}

// ---------------- launcher ----------------

extern "C" void kernel_launch(void* const* d_in, const int* in_sizes, int n_in,
                              void* d_out, int out_size, void* d_ws, size_t ws_size,
                              hipStream_t stream) {
  (void)in_sizes; (void)n_in; (void)out_size; (void)ws_size;

  const float* x    = (const float*)d_in[0];
  const float* cosp = (const float*)d_in[1];
  const float* sinp = (const float*)d_in[2];
  const float* wq   = (const float*)d_in[3];
  const float* wk   = (const float*)d_in[4];
  const float* wv   = (const float*)d_in[5];
  const float* wo   = (const float*)d_in[6];

  char* ws = (char*)d_ws;
  size_t off = 0;
  __bf16* Xb  = (__bf16*)(ws + off); off += (size_t)MTOT * E_ * 2;   // 16 MB
  __bf16* WqT = (__bf16*)(ws + off); off += (size_t)E_ * E_ * 2;     //  8 MB
  __bf16* WkT = (__bf16*)(ws + off); off += (size_t)E_ * E_ * 2;
  __bf16* WvT = (__bf16*)(ws + off); off += (size_t)E_ * E_ * 2;
  __bf16* WoT = (__bf16*)(ws + off); off += (size_t)E_ * E_ * 2;
  __bf16* Qb  = (__bf16*)(ws + off); off += (size_t)MTOT * E_ * 2;   // [B,H,S,D]
  __bf16* Kb  = (__bf16*)(ws + off); off += (size_t)MTOT * E_ * 2;
  __bf16* Vt  = (__bf16*)(ws + off); off += (size_t)MTOT * E_ * 2;   // [B,H,D,S]
  __bf16* AO  = (__bf16*)(ws + off); off += (size_t)MTOT * E_ * 2;   // [B,S,H*D]

  cvt_x_kernel<<<(MTOT * E_) / 256, 256, 0, stream>>>(x, Xb);
  cvt_wt_kernel<<<(E_ * E_) / 256, 256, 0, stream>>>(wq, WqT);
  cvt_wt_kernel<<<(E_ * E_) / 256, 256, 0, stream>>>(wk, WkT);
  cvt_wt_kernel<<<(E_ * E_) / 256, 256, 0, stream>>>(wv, WvT);
  cvt_wt_kernel<<<(E_ * E_) / 256, 256, 0, stream>>>(wo, WoT);

  qkv_rope_kernel<<<dim3(MTOT / 64, H_, 3), 256, 0, stream>>>(
      Xb, WqT, WkT, WvT, cosp, sinp, Qb, Kb, Vt);

  attn_kernel<<<dim3(B_ * H_, 16), 256, 0, stream>>>(Qb, Kb, Vt, AO);

  oproj_kernel<<<dim3(MTOT / 64, E_ / 128), 256, 0, stream>>>(AO, WoT, (float*)d_out);
}
